// GPT_32744830664997
// MI455X (gfx1250) — matrix operand order repaired and checked by reading, not compiled, not run on hardware
//
#include <hip/hip_runtime.h>
#include <hip/hip_bf16.h>
#include <math.h>

// ---------------------------------------------------------------------------
// GPT forward on gfx1250 (MI455X). ~255 GFLOP total, all weights fit in the
// 192MB L2 -> compute-bound -> all GEMMs through v_wmma_f32_16x16x32_bf16,
// with CDNA5 async global->LDS copies (ASYNCcnt) double-buffering the tiles.
// Block tile 128x128, 8 waves each computing a 32x64 sub-tile (2x4 WMMA).
// ---------------------------------------------------------------------------

typedef __attribute__((ext_vector_type(16))) __bf16 v16bf;
typedef __attribute__((ext_vector_type(8)))  __bf16 v8bf;
typedef __attribute__((ext_vector_type(8)))  float  v8f;

#define GL  4
#define GH  8
#define GD  512
#define GT  2048
#define GB  2
#define GV  32000
#define GHD 64
#define BT  (GB * GT)          // 4096 rows of activations

// ------------------------------- fp32 -> bf16 -------------------------------
__global__ void cvt_f32_to_bf16(const float* __restrict__ in,
                                __bf16* __restrict__ out, size_t n) {
  size_t i = (size_t)blockIdx.x * blockDim.x + threadIdx.x;
  size_t stride = (size_t)gridDim.x * blockDim.x;
  for (; i < n; i += stride) out[i] = (__bf16)in[i];
}

// --------------------- fp32 [R,C] -> bf16 transpose [C,R] -------------------
// Materialize wte^T once per call so the lm_head GEMM (~134 GFLOP, the
// largest) runs on the fast vectorized path.
__global__ __launch_bounds__(256)
void transpose_to_bf16(const float* __restrict__ in, __bf16* __restrict__ out,
                       int R, int C) {
  __shared__ float tile[32][33];
  int ctiles = C / 32;
  int bx = blockIdx.x % ctiles;   // column tile
  int by = blockIdx.x / ctiles;   // row tile
  int tx = threadIdx.x & 31;
  int ty = threadIdx.x >> 5;      // 0..7
  #pragma unroll
  for (int i = 0; i < 32; i += 8)
    tile[ty + i][tx] = in[(size_t)(by * 32 + ty + i) * C + bx * 32 + tx];
  __syncthreads();
  #pragma unroll
  for (int i = 0; i < 32; i += 8)
    out[(size_t)(bx * 32 + ty + i) * R + by * 32 + tx] = (__bf16)tile[tx][ty + i];
}

// ------------------------------- embedding ----------------------------------
__global__ void embed_kernel(const float* __restrict__ wte,
                             const float* __restrict__ wpe,
                             const int* __restrict__ idx,
                             float* __restrict__ x) {
  size_t n = (size_t)BT * GD;
  size_t stride = (size_t)gridDim.x * blockDim.x;
  for (size_t i = (size_t)blockIdx.x * blockDim.x + threadIdx.x; i < n; i += stride) {
    int d  = (int)(i % GD);
    int bt = (int)(i / GD);
    int t  = bt % GT;
    x[i] = wte[(size_t)idx[bt] * GD + d] + wpe[(size_t)t * GD + d];
  }
}

// ------------------------- LayerNorm (fp32 in, bf16 out) --------------------
__global__ __launch_bounds__(256)
void layernorm_bf16(const float* __restrict__ x, const float* __restrict__ w,
                    const float* __restrict__ b, __bf16* __restrict__ out) {
  int row = blockIdx.x;                // one row (D=512) per 256-thread block
  const float* xr = x + (size_t)row * GD;
  int t = threadIdx.x;
  float v0 = xr[t], v1 = xr[t + 256];

  __shared__ float red[8];
  float s = v0 + v1;
  #pragma unroll
  for (int off = 16; off > 0; off >>= 1) s += __shfl_xor(s, off);
  if ((t & 31) == 0) red[t >> 5] = s;
  __syncthreads();
  float mu = 0.f;
  #pragma unroll
  for (int i = 0; i < 8; ++i) mu += red[i];
  mu *= (1.0f / GD);

  float d0 = v0 - mu, d1 = v1 - mu;
  __syncthreads();
  s = d0 * d0 + d1 * d1;
  #pragma unroll
  for (int off = 16; off > 0; off >>= 1) s += __shfl_xor(s, off);
  if ((t & 31) == 0) red[t >> 5] = s;
  __syncthreads();
  float var = 0.f;
  #pragma unroll
  for (int i = 0; i < 8; ++i) var += red[i];
  var *= (1.0f / GD);
  float rstd = rsqrtf(var + 1e-5f);

  out[(size_t)row * GD + t]       = (__bf16)(d0 * rstd * w[t]       + b[t]);
  out[(size_t)row * GD + t + 256] = (__bf16)(d1 * rstd * w[t + 256] + b[t + 256]);
}

// -------------------- CDNA5 async global->LDS (ASYNCcnt) --------------------
__device__ __forceinline__ void async_copy_b128(void* lds, const void* gptr) {
  uint32_t l = (uint32_t)(size_t)lds;   // LDS byte address = low 32 addr bits
  asm volatile("global_load_async_to_lds_b128 %0, %1, off"
               :: "v"(l), "v"(gptr) : "memory");
}
__device__ __forceinline__ void wait_async0() {
  asm volatile("s_wait_asynccnt 0x0" ::: "memory");
}

// ------------------------------- WMMA GEMM ----------------------------------
// C[M,N] = act(A_bf16[M,K] @ B_bf16[K,N] + bias) (+ residual)
// flags bit1: exact GELU epilogue
// Block tile 128(M) x 128(N), BK=32, double-buffered async LDS staging (32KB).
// 8 waves in 4(M) x 2(N); each wave: 32x64 = 2x4 v_wmma_f32_16x16x32_bf16.
#define BM 128
#define BN 128
#define BK 32

__global__ __launch_bounds__(256)
void gemm_bf16_wmma(const __bf16* __restrict__ A, const __bf16* __restrict__ B,
                    const float* __restrict__ bias,
                    const float* __restrict__ residual,
                    float* __restrict__ Cf, __bf16* __restrict__ Cb,
                    int M, int N, int K, int flags) {
  __shared__ __align__(32) __bf16 As[2][BM][BK];   // 2 x 8 KB
  __shared__ __align__(32) __bf16 Bs[2][BK][BN];   // 2 x 8 KB

  const int tid  = threadIdx.x;
  const int lane = tid & 31;
  const int wave = tid >> 5;
  const int m0 = blockIdx.y * BM;
  const int n0 = blockIdx.x * BN;
  const int wm = (wave & 3) * 32;   // wave M offset in tile (4 groups)
  const int wn = (wave >> 2) * 64;  // wave N offset in tile (2 groups)

  // per-thread staging assignment (two b128 for A, two for B, per tile)
  const int ar = tid >> 2;              // A rows ar, ar+64
  const int ac = (tid & 3) * 8;         // 0,8,16,24
  const int br = tid >> 4;              // B rows br, br+16
  const int bc = (tid & 15) * 8;        // 0..120

  v8f c[2][4] = {};
  const int nk = K / BK;

  // prologue: fill buffer 0
  {
    async_copy_b128(&As[0][ar][ac],      &A[(size_t)(m0 + ar) * K + ac]);
    async_copy_b128(&As[0][ar + 64][ac], &A[(size_t)(m0 + ar + 64) * K + ac]);
    async_copy_b128(&Bs[0][br][bc],      &B[(size_t)br * N + n0 + bc]);
    async_copy_b128(&Bs[0][br + 16][bc], &B[(size_t)(br + 16) * N + n0 + bc]);
  }

  for (int kt = 0; kt < nk; ++kt) {
    const int cur = kt & 1;
    wait_async0();       // copies for buffer `cur` complete (per wave)
    __syncthreads();     // whole block sees the staged tile

    if (kt + 1 < nk) {   // kick next tile into the other buffer (overlaps WMMA)
      const int k1 = (kt + 1) * BK;
      const int nxt = cur ^ 1;
      async_copy_b128(&As[nxt][ar][ac],      &A[(size_t)(m0 + ar) * K + k1 + ac]);
      async_copy_b128(&As[nxt][ar + 64][ac], &A[(size_t)(m0 + ar + 64) * K + k1 + ac]);
      async_copy_b128(&Bs[nxt][br][bc],      &B[(size_t)(k1 + br) * N + n0 + bc]);
      async_copy_b128(&Bs[nxt][br + 16][bc], &B[(size_t)(k1 + br + 16) * N + n0 + bc]);
    }

    // ---- fragments (layouts per CDNA5 ISA 7.12.2) ----
    // A 16x32 bf16: lane&15 = M row; lane>=16 takes K groups {8..15,24..31}
    int am = lane & 15;
    int kh = (lane >> 4) * 8;
    v16bf a[2];
    #pragma unroll
    for (int fm = 0; fm < 2; ++fm) {
      v8bf lo = *(const v8bf*)&As[cur][wm + fm * 16 + am][kh];
      v8bf hi = *(const v8bf*)&As[cur][wm + fm * 16 + am][16 + kh];
      #pragma unroll
      for (int j = 0; j < 8; ++j) { a[fm][j] = lo[j]; a[fm][8 + j] = hi[j]; }
    }
    // B 32x16 bf16: lane&15 = N column; K groups same as A
    v16bf bfr[4];
    #pragma unroll
    for (int fn = 0; fn < 4; ++fn) {
      #pragma unroll
      for (int j = 0; j < 8; ++j) {
        bfr[fn][j]     = Bs[cur][kh + j][wn + fn * 16 + am];
        bfr[fn][8 + j] = Bs[cur][16 + kh + j][wn + fn * 16 + am];
      }
    }

    #pragma unroll
    for (int fm = 0; fm < 2; ++fm) {
      #pragma unroll
      for (int fn = 0; fn < 4; ++fn) {
        c[fm][fn] = __builtin_amdgcn_wmma_f32_16x16x32_bf16(
            false, a[fm], false, bfr[fn], (short)0, c[fm][fn], false, false);
      }
    }
    // Next iteration's top barrier guarantees all waves finished reading
    // buffer `cur` before anyone's async copies overwrite it (distance 2).
  }

  // ---- epilogue: C/D layout: VGPR r -> M = r + 8*(lane>=16), N = lane&15 ----
  const bool do_gelu = (flags & 2) != 0;
  int nl = lane & 15;
  int mh = (lane >> 4) * 8;
  #pragma unroll
  for (int fm = 0; fm < 2; ++fm) {
    #pragma unroll
    for (int fn = 0; fn < 4; ++fn) {
      #pragma unroll
      for (int r = 0; r < 8; ++r) {
        int gm = m0 + wm + fm * 16 + mh + r;
        int gn = n0 + wn + fn * 16 + nl;
        float v = c[fm][fn][r];
        if (bias) v += bias[gn];
        if (do_gelu) v = 0.5f * v * (1.0f + erff(v * 0.70710678118f));
        if (residual) v += residual[(size_t)gm * N + gn];
        size_t o = (size_t)gm * N + gn;
        if (Cf) Cf[o] = v;
        if (Cb) Cb[o] = (__bf16)v;
      }
    }
  }
}

// --------------------------- causal attention -------------------------------
// One wave per query row; lanes split head-dim (2 fp32 each); online softmax,
// 4 keys per iteration so the 4 shuffle-reduce chains overlap (ILP) and only
// one exp-rescale happens per group of 4 keys.
__global__ __launch_bounds__(256)
void attention_kernel(const float* __restrict__ qkv, __bf16* __restrict__ y) {
  int wid  = blockIdx.x * 8 + (threadIdx.x >> 5);   // global wave id
  int lane = threadIdx.x & 31;
  int tq = wid % GT;
  int bh = wid / GT;
  int h  = bh % GH;
  int b  = bh / GH;

  const size_t row3d = (size_t)3 * GD;
  const size_t base  = (size_t)b * GT * row3d;
  const float* qp = qkv + base + (size_t)tq * row3d + h * GHD;
  int d0 = lane * 2;
  float q0 = qp[d0], q1 = qp[d0 + 1];

  float m = -__builtin_inff(), l = 0.f, a0 = 0.f, a1 = 0.f;
  const float scale = 0.125f;                        // 1/sqrt(64)

  const int nkeys = tq + 1;
  const int n4 = nkeys & ~3;
  int t = 0;
  for (; t < n4; t += 4) {
    const float* k0p = qkv + base + (size_t)(t    ) * row3d + GD + h * GHD;
    const float* k1p = qkv + base + (size_t)(t + 1) * row3d + GD + h * GHD;
    const float* k2p = qkv + base + (size_t)(t + 2) * row3d + GD + h * GHD;
    const float* k3p = qkv + base + (size_t)(t + 3) * row3d + GD + h * GHD;
    float s0 = q0 * k0p[d0] + q1 * k0p[d0 + 1];
    float s1 = q0 * k1p[d0] + q1 * k1p[d0 + 1];
    float s2 = q0 * k2p[d0] + q1 * k2p[d0 + 1];
    float s3 = q0 * k3p[d0] + q1 * k3p[d0 + 1];
    #pragma unroll
    for (int off = 16; off > 0; off >>= 1) {
      s0 += __shfl_xor(s0, off);
      s1 += __shfl_xor(s1, off);
      s2 += __shfl_xor(s2, off);
      s3 += __shfl_xor(s3, off);
    }
    s0 *= scale; s1 *= scale; s2 *= scale; s3 *= scale;
    float mg = fmaxf(fmaxf(s0, s1), fmaxf(s2, s3));
    float mn = fmaxf(m, mg);
    float corr = __expf(m - mn);
    float p0 = __expf(s0 - mn), p1 = __expf(s1 - mn);
    float p2 = __expf(s2 - mn), p3 = __expf(s3 - mn);
    const float* v0p = qkv + base + (size_t)(t    ) * row3d + 2 * GD + h * GHD;
    const float* v1p = qkv + base + (size_t)(t + 1) * row3d + 2 * GD + h * GHD;
    const float* v2p = qkv + base + (size_t)(t + 2) * row3d + 2 * GD + h * GHD;
    const float* v3p = qkv + base + (size_t)(t + 3) * row3d + 2 * GD + h * GHD;
    a0 = a0 * corr + p0 * v0p[d0]     + p1 * v1p[d0]     + p2 * v2p[d0]     + p3 * v3p[d0];
    a1 = a1 * corr + p0 * v0p[d0 + 1] + p1 * v1p[d0 + 1] + p2 * v2p[d0 + 1] + p3 * v3p[d0 + 1];
    l  = l * corr + (p0 + p1 + p2 + p3);
    m  = mn;
  }
  for (; t < nkeys; ++t) {
    const float* kp = qkv + base + (size_t)t * row3d + GD + h * GHD;
    float s = q0 * kp[d0] + q1 * kp[d0 + 1];
    #pragma unroll
    for (int off = 16; off > 0; off >>= 1) s += __shfl_xor(s, off);
    s *= scale;
    float mn   = fmaxf(m, s);
    float corr = __expf(m - mn);
    float p    = __expf(s - mn);
    const float* vp = qkv + base + (size_t)t * row3d + 2 * GD + h * GHD;
    a0 = a0 * corr + p * vp[d0];
    a1 = a1 * corr + p * vp[d0 + 1];
    l  = l * corr + p;
    m  = mn;
  }
  float inv = 1.0f / l;
  __bf16* yp = y + ((size_t)(b * GT + tq)) * GD + h * GHD;
  yp[d0]     = (__bf16)(a0 * inv);
  yp[d0 + 1] = (__bf16)(a1 * inv);
}

// ------------------------------- orchestration ------------------------------
extern "C" void kernel_launch(void* const* d_in, const int* in_sizes, int n_in,
                              void* d_out, int out_size, void* d_ws, size_t ws_size,
                              hipStream_t stream) {
  const float* wte    = (const float*)d_in[0];
  const float* wpe    = (const float*)d_in[1];
  const float* ln1_w  = (const float*)d_in[2];
  const float* ln1_b  = (const float*)d_in[3];
  const float* attn_w = (const float*)d_in[4];
  const float* attn_b = (const float*)d_in[5];
  const float* proj_w = (const float*)d_in[6];
  const float* proj_b = (const float*)d_in[7];
  const float* ln2_w  = (const float*)d_in[8];
  const float* ln2_b  = (const float*)d_in[9];
  const float* fc_w   = (const float*)d_in[10];
  const float* fc_b   = (const float*)d_in[11];
  const float* fc2_w  = (const float*)d_in[12];
  const float* fc2_b  = (const float*)d_in[13];
  const float* lnf_w  = (const float*)d_in[14];
  const float* lnf_b  = (const float*)d_in[15];
  const int*   idx    = (const int*)d_in[16];
  float* out = (float*)d_out;

  // ---- workspace layout (256B aligned slabs) ----
  char* ws = (char*)d_ws;
  size_t off = 0;
  auto alloc = [&](size_t bytes) -> char* {
    char* p = ws + off;
    off = (off + bytes + 255) & ~(size_t)255;
    return p;
  };
  __bf16* wteT_bf = (__bf16*)alloc((size_t)GD * GV * 2);       // wte^T [D,V]
  __bf16* aw_bf   = (__bf16*)alloc((size_t)GL * GD * 3 * GD * 2);
  __bf16* pw_bf   = (__bf16*)alloc((size_t)GL * GD * GD * 2);
  __bf16* fw_bf   = (__bf16*)alloc((size_t)GL * GD * 4 * GD * 2);
  __bf16* f2w_bf  = (__bf16*)alloc((size_t)GL * 4 * GD * GD * 2);
  float*  x       = (float*)alloc((size_t)BT * GD * 4);
  __bf16* hbf     = (__bf16*)alloc((size_t)BT * GD * 2);
  float*  qkv     = (float*)alloc((size_t)BT * 3 * GD * 4);
  __bf16* ybf     = (__bf16*)alloc((size_t)BT * GD * 2);
  __bf16* h4bf    = (__bf16*)alloc((size_t)BT * 4 * GD * 2);
  (void)ws_size; (void)n_in; (void)in_sizes; (void)out_size;

  // ---- weights to bf16 (deterministic, redone every call) ----
  transpose_to_bf16<<<(GD / 32) * (GV / 32), 256, 0, stream>>>(wte, wteT_bf, GV, GD);
  cvt_f32_to_bf16<<<1024, 256, 0, stream>>>(attn_w, aw_bf,  (size_t)GL * GD * 3 * GD);
  cvt_f32_to_bf16<<<512,  256, 0, stream>>>(proj_w, pw_bf,  (size_t)GL * GD * GD);
  cvt_f32_to_bf16<<<1024, 256, 0, stream>>>(fc_w,   fw_bf,  (size_t)GL * GD * 4 * GD);
  cvt_f32_to_bf16<<<1024, 256, 0, stream>>>(fc2_w,  f2w_bf, (size_t)GL * 4 * GD * GD);

  // ---- embedding ----
  embed_kernel<<<8192, 256, 0, stream>>>(wte, wpe, idx, x);

  auto gemm = [&](const __bf16* A, const __bf16* B, const float* bias,
                  const float* residual, float* Cf, __bf16* Cb,
                  int M, int N, int K, int flags) {
    dim3 grid(N / BN, M / BM);
    gemm_bf16_wmma<<<grid, 256, 0, stream>>>(A, B, bias, residual, Cf, Cb,
                                             M, N, K, flags);
  };

  for (int l = 0; l < GL; ++l) {
    // LN1 -> bf16
    layernorm_bf16<<<BT, 256, 0, stream>>>(x, ln1_w + l * GD, ln1_b + l * GD, hbf);
    // QKV: [4096,512] x [512,1536]
    gemm(hbf, aw_bf + (size_t)l * GD * 3 * GD, attn_b + l * 3 * GD,
         nullptr, qkv, nullptr, BT, 3 * GD, GD, 0);
    // attention -> y (bf16)
    attention_kernel<<<(GB * GH * GT) / 8, 256, 0, stream>>>(qkv, ybf);
    // proj + residual: x = x + y @ pw + pb
    gemm(ybf, pw_bf + (size_t)l * GD * GD, proj_b + l * GD,
         x, x, nullptr, BT, GD, GD, 0);
    // LN2 -> bf16
    layernorm_bf16<<<BT, 256, 0, stream>>>(x, ln2_w + l * GD, ln2_b + l * GD, hbf);
    // FC + GELU -> bf16 h4
    gemm(hbf, fw_bf + (size_t)l * GD * 4 * GD, fc_b + l * 4 * GD,
         nullptr, nullptr, h4bf, BT, 4 * GD, GD, 2);
    // FC2 + residual: x = x + gelu(h2) @ f2w + f2b
    gemm(h4bf, f2w_bf + (size_t)l * 4 * GD * GD, fc2_b + l * GD,
         x, x, nullptr, BT, GD, 4 * GD, 0);
  }

  // final LN -> bf16
  layernorm_bf16<<<BT, 256, 0, stream>>>(x, lnf_w, lnf_b, hbf);
  // lm_head: logits = h @ wte^T  (pre-transposed bf16, fast path)
  gemm(hbf, wteT_bf, nullptr, nullptr, out, nullptr, BT, GV, GD, 0);
}
